// LSTM_7078106104556
// MI455X (gfx1250) — compile-verified, matrix operand order checked by reading, not run on hardware
//
#include <hip/hip_runtime.h>

#define SEQ   512
#define BATCH 64
#define NHID  1024
#define FOURH 4096
#define SB    (SEQ * BATCH)   // 32768

typedef __bf16 bf16;
typedef bf16  v16bf __attribute__((ext_vector_type(16)));
typedef bf16  v8bf  __attribute__((ext_vector_type(8)));
typedef float v8f   __attribute__((ext_vector_type(8)));

__device__ __forceinline__ bf16 f2bf(float f) {
    unsigned u = __builtin_bit_cast(unsigned, f);
    u += 0x7FFFu + ((u >> 16) & 1u);            // round-to-nearest-even
    unsigned short h = (unsigned short)(u >> 16);
    return __builtin_bit_cast(bf16, h);
}

__device__ __forceinline__ float sigmoidf_(float x) {
    return 1.0f / (1.0f + __expf(-x));
}

// Load one 16x32 bf16 WMMA A/B fragment from a row-major matrix.
// row must already include (lane&15); per ISA layout lane L holds row (L&15),
// K chunks at k0+(L>>4)*8 and +16 (two 16-byte contiguous loads).
__device__ __forceinline__ v16bf load_frag(const bf16* __restrict__ base,
                                           int row, int ld, int k0, int lane) {
    const bf16* p = base + (long)row * ld + k0 + ((lane >> 4) << 3);
    v8bf lo = *(const v8bf*)(p);
    v8bf hi = *(const v8bf*)(p + 16);
    v16bf r;
#pragma unroll
    for (int i = 0; i < 8; ++i) { r[i] = lo[i]; r[8 + i] = hi[i]; }
    return r;
}

// ---------------- elementwise helpers ----------------
__global__ void cvt_f32_to_bf16(const float* __restrict__ in, bf16* __restrict__ out, long n) {
    long i = (long)blockIdx.x * blockDim.x + threadIdx.x;
    long s = (long)gridDim.x * blockDim.x;
    for (; i < n; i += s) out[i] = f2bf(in[i]);
}

__global__ void zero_u32(unsigned* __restrict__ p, int n) {
    int i = blockIdx.x * blockDim.x + threadIdx.x;
    if (i < n) p[i] = 0u;
}

// ---------------- generic GEMM: Out = X @ W^T (+bias) ----------------
// X: [M,K] bf16 row-major, W: [N,K] bf16 row-major.
// Each wave computes a 32x32 tile with 4 wmma accumulators.
// Writes fp32 (OutF) and/or bf16 (OutB). M,N multiples of 32; K multiple of 32.
__global__ void gemm_xwT(const bf16* __restrict__ X, const bf16* __restrict__ W,
                         const float* __restrict__ bias,
                         float* __restrict__ OutF, bf16* __restrict__ OutB,
                         int M, int N, int K) {
    const int lane = threadIdx.x & 31;
    const int wid  = blockIdx.x * (blockDim.x >> 5) + (threadIdx.x >> 5);
    const int tilesN = N >> 5;
    const int tm = wid / tilesN;
    const int tn = wid % tilesN;
    if (tm * 32 >= M) return;
    const int r = lane & 15;

    v8f acc00 = {}, acc01 = {}, acc10 = {}, acc11 = {};
    for (int k0 = 0; k0 < K; k0 += 32) {
        v16bf a0 = load_frag(X, tm * 32 + r,      K, k0, lane);
        v16bf a1 = load_frag(X, tm * 32 + 16 + r, K, k0, lane);
        v16bf b0 = load_frag(W, tn * 32 + r,      K, k0, lane);
        v16bf b1 = load_frag(W, tn * 32 + 16 + r, K, k0, lane);
        acc00 = __builtin_amdgcn_wmma_f32_16x16x32_bf16(false, a0, false, b0, (short)0, acc00, false, false);
        acc01 = __builtin_amdgcn_wmma_f32_16x16x32_bf16(false, a0, false, b1, (short)0, acc01, false, false);
        acc10 = __builtin_amdgcn_wmma_f32_16x16x32_bf16(false, a1, false, b0, (short)0, acc10, false, false);
        acc11 = __builtin_amdgcn_wmma_f32_16x16x32_bf16(false, a1, false, b1, (short)0, acc11, false, false);
    }

    const int mrb = (lane >> 4) << 3;   // 0 or 8
#pragma unroll
    for (int mi = 0; mi < 2; ++mi) {
        const v8f& acn0 = mi ? acc10 : acc00;
        const v8f& acn1 = mi ? acc11 : acc01;
#pragma unroll
        for (int ni = 0; ni < 2; ++ni) {
            const v8f& a = ni ? acn1 : acn0;
            const int n  = tn * 32 + ni * 16 + r;
            const float bv = bias ? bias[n] : 0.0f;
#pragma unroll
            for (int e = 0; e < 8; ++e) {
                const int  m   = tm * 32 + mi * 16 + mrb + e;
                const long idx = (long)m * N + n;
                const float v  = a[e] + bv;
                if (OutF) OutF[idx] = v;
                if (OutB) OutB[idx] = f2bf(v);
            }
        }
    }
}

// ---------------- persistent LSTM layer (all 512 steps, one launch) ----------------
// Grid MUST be 32 blocks x 256 threads = 256 waves = 4 m-tiles x 64 n-tiles.
// Each wave owns a 16(M)x16(N) slice and all four gates for it; the cell state c
// for that slice lives in registers for the entire sequence. Steps are separated
// by a device-scope atomic counter barrier (monotonic: target = 32*(t+1)).
__global__ void lstm_layer(const float* __restrict__ Apre,   // [SEQ, 64, 4096] x-part + bias
                           const bf16*  __restrict__ Wh,     // [4096, 1024]
                           bf16*  __restrict__ Hall,         // [SEQ, 64, 1024] h_t out (bf16)
                           float* __restrict__ hOut,         // [64, 1024] final h (fp32)
                           float* __restrict__ cOut,         // [64, 1024] final c (fp32)
                           unsigned* __restrict__ bar) {
    const int lane = threadIdx.x & 31;
    const int wid  = blockIdx.x * (blockDim.x >> 5) + (threadIdx.x >> 5);
    const int tm = wid >> 6;   // 0..3
    const int tn = wid & 63;   // 0..63
    const int r  = lane & 15;

    const int n  = tn * 16 + r;
    const int mb = tm * 16 + ((lane >> 4) << 3);

    float c_reg[8];
#pragma unroll
    for (int e = 0; e < 8; ++e) c_reg[e] = 0.0f;

    for (int t = 0; t < SEQ; ++t) {
        v8f acc_i = {}, acc_f = {}, acc_o = {}, acc_g = {};
        if (t > 0) {                         // h_{-1} == 0: skip the GEMM at t==0
            const bf16* hprev = Hall + (size_t)(t - 1) * BATCH * NHID;
            for (int k0 = 0; k0 < NHID; k0 += 32) {
                v16bf a  = load_frag(hprev, tm * 16 + r, NHID, k0, lane);
                v16bf bi = load_frag(Wh, 0 * NHID + tn * 16 + r, NHID, k0, lane);
                v16bf bf = load_frag(Wh, 1 * NHID + tn * 16 + r, NHID, k0, lane);
                v16bf bo = load_frag(Wh, 2 * NHID + tn * 16 + r, NHID, k0, lane);
                v16bf bg = load_frag(Wh, 3 * NHID + tn * 16 + r, NHID, k0, lane);
                acc_i = __builtin_amdgcn_wmma_f32_16x16x32_bf16(false, a, false, bi, (short)0, acc_i, false, false);
                acc_f = __builtin_amdgcn_wmma_f32_16x16x32_bf16(false, a, false, bf, (short)0, acc_f, false, false);
                acc_o = __builtin_amdgcn_wmma_f32_16x16x32_bf16(false, a, false, bo, (short)0, acc_o, false, false);
                acc_g = __builtin_amdgcn_wmma_f32_16x16x32_bf16(false, a, false, bg, (short)0, acc_g, false, false);
            }
        }

        const float* At = Apre + (size_t)t * BATCH * FOURH;
        bf16* ht = Hall + (size_t)t * BATCH * NHID;
#pragma unroll
        for (int e = 0; e < 8; ++e) {
            const int  m    = mb + e;
            const long abse = (long)m * FOURH + n;
            const float ai = acc_i[e] + At[abse];
            const float af = acc_f[e] + At[abse + 1024];
            const float ao = acc_o[e] + At[abse + 2048];
            const float ag = acc_g[e] + At[abse + 3072];
            const float iv = sigmoidf_(ai);
            const float fv = sigmoidf_(af);
            const float ov = sigmoidf_(ao);
            const float gv = tanhf(ag);
            const float cn = fv * c_reg[e] + iv * gv;
            const float hn = ov * tanhf(cn);
            c_reg[e] = cn;
            const long hidx = (long)m * NHID + n;
            ht[hidx] = f2bf(hn);
            if (t == SEQ - 1) {
                hOut[hidx] = hn;
                cOut[hidx] = cn;
            }
        }

        // ---- device-scope step barrier ----
        __threadfence();                     // release this block's h stores
        __syncthreads();
        if (threadIdx.x == 0) {
            __hip_atomic_fetch_add(bar, 1u, __ATOMIC_RELEASE, __HIP_MEMORY_SCOPE_AGENT);
            const unsigned target = (unsigned)(t + 1) * gridDim.x;
            while (__hip_atomic_load(bar, __ATOMIC_ACQUIRE, __HIP_MEMORY_SCOPE_AGENT) < target)
                __builtin_amdgcn_s_sleep(2);
        }
        __syncthreads();
        __threadfence();                     // acquire: invalidate stale lines in every wave's cache path
    }
}

// ---------------- host side ----------------
extern "C" void kernel_launch(void* const* d_in, const int* in_sizes, int n_in,
                              void* d_out, int out_size, void* d_ws, size_t ws_size,
                              hipStream_t stream) {
    const float* x   = (const float*)d_in[0];
    const float* wx0 = (const float*)d_in[1];
    const float* wh0 = (const float*)d_in[2];
    const float* b0  = (const float*)d_in[3];
    const float* wo0 = (const float*)d_in[4];
    const float* wx1 = (const float*)d_in[5];
    const float* wh1 = (const float*)d_in[6];
    const float* b1  = (const float*)d_in[7];
    const float* wo1 = (const float*)d_in[8];
    float* out = (float*)d_out;

    // workspace carve-up (256B aligned)
    char*  ws  = (char*)d_ws;
    size_t off = 0;
    auto carve = [&](size_t bytes) -> void* {
        off = (off + 255) & ~(size_t)255;
        void* p = ws + off;
        off += bytes;
        return p;
    };
    float*    A    = (float*)   carve((size_t)SB * FOURH * 4);  // 512 MiB preactivations
    bf16*     xb   = (bf16*)    carve((size_t)SB * NHID * 2);   // x in bf16
    bf16*     Hall = (bf16*)    carve((size_t)SB * NHID * 2);   // per-step h storage (reused per layer)
    bf16*     p0b  = (bf16*)    carve((size_t)SB * NHID * 2);   // layer-0 projected output
    bf16*     wx0b = (bf16*)    carve((size_t)FOURH * NHID * 2);
    bf16*     wh0b = (bf16*)    carve((size_t)FOURH * NHID * 2);
    bf16*     wo0b = (bf16*)    carve((size_t)NHID * NHID * 2);
    bf16*     wx1b = (bf16*)    carve((size_t)FOURH * NHID * 2);
    bf16*     wh1b = (bf16*)    carve((size_t)FOURH * NHID * 2);
    bf16*     wo1b = (bf16*)    carve((size_t)NHID * NHID * 2);
    unsigned* bars = (unsigned*)carve(2 * sizeof(unsigned));    // step-barrier counters

    const int  T  = 256;
    const long HC = (long)BATCH * NHID;                 // 65536
    float* tail = out + (size_t)SEQ * BATCH * NHID;     // [h0 | h1 | c0 | c1]

    // fp32 -> bf16 conversions
    cvt_f32_to_bf16<<<4096, T, 0, stream>>>(x,   xb,   (long)SB * NHID);
    cvt_f32_to_bf16<<<2048, T, 0, stream>>>(wx0, wx0b, (long)FOURH * NHID);
    cvt_f32_to_bf16<<<2048, T, 0, stream>>>(wh0, wh0b, (long)FOURH * NHID);
    cvt_f32_to_bf16<<<1024, T, 0, stream>>>(wo0, wo0b, (long)NHID * NHID);
    cvt_f32_to_bf16<<<2048, T, 0, stream>>>(wx1, wx1b, (long)FOURH * NHID);
    cvt_f32_to_bf16<<<2048, T, 0, stream>>>(wh1, wh1b, (long)FOURH * NHID);
    cvt_f32_to_bf16<<<1024, T, 0, stream>>>(wo1, wo1b, (long)NHID * NHID);
    zero_u32<<<1, 32, 0, stream>>>(bars, 2);

    const int gemmA_blocks = (SB / 32) * (FOURH / 32) / 8;  // 16384
    const int gemmP_blocks = (SB / 32) * (NHID  / 32) / 8;  // 4096

    // ----- layer 0 -----
    gemm_xwT<<<gemmA_blocks, T, 0, stream>>>(xb, wx0b, b0, A, (bf16*)nullptr, SB, FOURH, NHID);
    lstm_layer<<<32, T, 0, stream>>>(A, wh0b, Hall, tail + 0 * HC, tail + 2 * HC, bars + 0);
    // batched projection of ALL timesteps: p0 = Hall @ wo0^T  (bf16, feeds layer 1)
    gemm_xwT<<<gemmP_blocks, T, 0, stream>>>(Hall, wo0b, (const float*)nullptr,
                                             (float*)nullptr, p0b, SB, NHID, NHID);

    // ----- layer 1 -----
    gemm_xwT<<<gemmA_blocks, T, 0, stream>>>(p0b, wx1b, b1, A, (bf16*)nullptr, SB, FOURH, NHID);
    lstm_layer<<<32, T, 0, stream>>>(A, wh1b, Hall, tail + 1 * HC, tail + 3 * HC, bars + 1);
    // batched projection of ALL timesteps: out = Hall @ wo1^T  (fp32 final output)
    gemm_xwT<<<gemmP_blocks, T, 0, stream>>>(Hall, wo1b, (const float*)nullptr,
                                             out, (bf16*)nullptr, SB, NHID, NHID);
}
